// LongformerTripleClassifier_16011638080096
// MI455X (gfx1250) — matrix operand order repaired and checked by reading, not compiled, not run on hardware
//
#include <hip/hip_runtime.h>
#include <math.h>

#define SQ   4096
#define HID  768
#define NHD  12
#define DH   64
#define LAYERS 12
#define FFD  3072
#define WINW 256
#define NGT  16
#define NEGF (-1.0e9f)

typedef __bf16 bf16_t;
typedef bf16_t v16bf __attribute__((ext_vector_type(16)));
typedef float  v8f   __attribute__((ext_vector_type(8)));
typedef unsigned int v4u __attribute__((ext_vector_type(4)));
typedef int  v8i __attribute__((ext_vector_type(8)));
typedef int  v4i __attribute__((ext_vector_type(4)));

union FragB16 { unsigned int u[8]; v16bf v; };

__device__ __forceinline__ unsigned short f2bf(float f) {
  unsigned int u = __builtin_bit_cast(unsigned int, f);
  unsigned int r = u + 0x7FFFu + ((u >> 16) & 1u);
  return (unsigned short)(r >> 16);
}
__device__ __forceinline__ unsigned int packbf(float a, float b) {
  return (unsigned int)f2bf(a) | ((unsigned int)f2bf(b) << 16);
}
__device__ __forceinline__ float bf2f(unsigned short u) {
  unsigned int i = ((unsigned int)u) << 16;
  return __builtin_bit_cast(float, i);
}

// ---------------------------------------------------------------------------
// TDM: async 2D tile load (bf16) global -> LDS.  Tile = nrows lines of 32
// elements, line stride = linelen elements.  LDS written linearly (row-major
// [row][32]) which matches the WMMA fragment staging layouts below.
// D# per CDNA5 ISA ch.8: group0 = {count=1, lds_addr, global_addr, type=2},
// group1 = {data_size=2B, tensor_dim0=linelen, tensor_dim1=nrows,
//           tile_dim0=32, tile_dim1=nrows, dim0_stride=linelen}.
// This toolchain exposes the 6-arg builtin (g0, g1, g2, g3, g4, cpol).
// ---------------------------------------------------------------------------
__device__ __forceinline__ void tdm_load_tile(unsigned lds_byte_off,
                                              const unsigned short* gsrc,
                                              int linelen, int nrows) {
  unsigned long long ga = (unsigned long long)(size_t)gsrc;
  v4u g0; v8i g1; v4i gz4 = {0, 0, 0, 0}; v8i gz8 = {0, 0, 0, 0, 0, 0, 0, 0};
  g0[0] = 1u;                                                   // count=1 (valid)
  g0[1] = lds_byte_off;                                         // LDS address
  g0[2] = (unsigned)ga;                                         // global addr lo
  g0[3] = ((unsigned)(ga >> 32) & 0x01FFFFFFu) | 0x80000000u;   // addr hi | type=2
  g1[0] = (int)(1u << 16);                                      // data_size = 2B
  g1[1] = (int)(((unsigned)linelen & 0xFFFFu) << 16);           // tensor_dim0 lo16
  g1[2] = (int)((((unsigned)linelen) >> 16) |
                (((unsigned)nrows & 0xFFFFu) << 16));           // dim0 hi | dim1 lo
  g1[3] = (int)((((unsigned)nrows) >> 16) | (32u << 16));       // dim1 hi | tile_dim0=32
  g1[4] = (int)(unsigned)nrows;                                 // tile_dim1 (tile_dim2=0)
  g1[5] = (int)(unsigned)linelen;                               // dim0_stride lo32
  g1[6] = 0;
  g1[7] = 0;
  __builtin_amdgcn_tensor_load_to_lds(g0, g1, gz4, gz4, gz8, 0);
}

// ---------------------------------------------------------------------------
// Weight convert+transpose: W[K][N] fp32 -> Wt[N][K] bf16 (per layer, L2 hot)
// ---------------------------------------------------------------------------
__global__ __launch_bounds__(256)
void transpose_f32_bf16_kernel(const float* __restrict__ W, unsigned short* __restrict__ Wt,
                               int K, int N) {
  __shared__ float t[32][33];
  const int kb = blockIdx.x * 32, nb = blockIdx.y * 32;
  for (int j = threadIdx.y; j < 32; j += 8)
    t[j][threadIdx.x] = W[(size_t)(kb + j) * N + nb + threadIdx.x];
  __syncthreads();
  for (int j = threadIdx.y; j < 32; j += 8)
    Wt[(size_t)(nb + j) * K + kb + threadIdx.x] = f2bf(t[threadIdx.x][j]);
}

// ---------------------------------------------------------------------------
// bf16-WMMA GEMM with double-buffered TDM tile staging.
// out[M,N] = act(A[M,K] @ Wt[N,K]^T + bias) ; out fp32 or bf16.
// BM=128, BN=64, BK=32; 8 waves, each wave 2x2 16x16 WMMA tiles.
// ---------------------------------------------------------------------------
__global__ __launch_bounds__(256)
void gemm_bf16_kernel(const unsigned short* __restrict__ A,   // [M][K] bf16
                      const unsigned short* __restrict__ Wt,  // [N][K] bf16
                      const float* __restrict__ bias,
                      float* __restrict__ outF, unsigned short* __restrict__ outB,
                      int M, int N, int K, int act) {
  __shared__ unsigned short As[2][128 * 32];  // [row][k]
  __shared__ unsigned short Bs[2][64 * 32];   // [col][k]
  const int tid = threadIdx.x;
  const int lane = tid & 31;
  const int w = tid >> 5;
  const int wm = w & 3, wn = w >> 2;
  const int m = lane & 15, hi = lane >> 4;
  const int rowBase = blockIdx.y * 128;
  const int colBase = blockIdx.x * 64;
  const int nk = K >> 5;

  const unsigned ldsA0 = (unsigned)(size_t)&As[0][0];
  const unsigned ldsA1 = (unsigned)(size_t)&As[1][0];
  const unsigned ldsB0 = (unsigned)(size_t)&Bs[0][0];
  const unsigned ldsB1 = (unsigned)(size_t)&Bs[1][0];

  if (w == 0) {
    tdm_load_tile(ldsA0, A + (size_t)rowBase * K, K, 128);
    tdm_load_tile(ldsB0, Wt + (size_t)colBase * K, K, 64);
  }

  v8f acc[2][2] = {};
  for (int i = 0; i < nk; ++i) {
    const int buf = i & 1;
    if (w == 0) {
      if (i + 1 < nk) {
        tdm_load_tile(buf ? ldsA0 : ldsA1, A + (size_t)rowBase * K + (size_t)(i + 1) * 32, K, 128);
        tdm_load_tile(buf ? ldsB0 : ldsB1, Wt + (size_t)colBase * K + (size_t)(i + 1) * 32, K, 64);
        __builtin_amdgcn_s_wait_tensorcnt(2);   // current tile's 2 loads done
      } else {
        __builtin_amdgcn_s_wait_tensorcnt(0);
      }
    }
    __syncthreads();
    FragB16 af[2], bfr[2];
    for (int mi = 0; mi < 2; ++mi) {
      int row = wm * 32 + mi * 16 + m;
      for (int g = 0; g < 8; ++g) {
        int kk0 = ((g & 4) << 2) | ((g & 3) << 1) | (hi << 3);
        af[mi].u[g] = *(const unsigned int*)&As[buf][row * 32 + kk0];
      }
    }
    for (int ni = 0; ni < 2; ++ni) {
      int col = wn * 32 + ni * 16 + m;
      for (int g = 0; g < 8; ++g)
        bfr[ni].u[g] = *(const unsigned int*)&Bs[buf][col * 32 + (((g & 4) << 2) | ((g & 3) << 1) | (hi << 3))];
    }
    for (int mi = 0; mi < 2; ++mi)
      for (int ni = 0; ni < 2; ++ni)
        acc[mi][ni] = __builtin_amdgcn_wmma_f32_16x16x32_bf16(
            false, af[mi].v, false, bfr[ni].v, (short)0, acc[mi][ni], false, false);
    __syncthreads();
  }
  for (int mi = 0; mi < 2; ++mi)
    for (int ni = 0; ni < 2; ++ni)
      for (int r = 0; r < 8; ++r) {
        int row = rowBase + wm * 32 + mi * 16 + r + 8 * hi;
        int col = colBase + wn * 32 + ni * 16 + m;
        float v = acc[mi][ni][r] + bias[col];
        if (act == 1) v = 0.5f * v * (1.0f + erff(v * 0.70710678118f));
        if (outF) outF[(size_t)row * N + col] = v;
        else      outB[(size_t)row * N + col] = f2bf(v);
      }
}

// ---------------------------------------------------------------------------
// WMMA flash attention: sliding window (768 keys) + 16 global keys, bf16 QKV.
// Scale (DH^-0.5) folded into post-WMMA f32 scores.
// ---------------------------------------------------------------------------
__global__ __launch_bounds__(128)
void attn_local_kernel(const unsigned short* __restrict__ qkvb, const int* __restrict__ amask,
                       const int* __restrict__ gmask, const int* __restrict__ gidx,
                       const float* __restrict__ gvalid, unsigned short* __restrict__ atto_bf) {
  __shared__ unsigned short Kt[32 * 64];    // [key][d]
  __shared__ unsigned short Vt[64 * 32];    // [d][key]
  __shared__ float pl[4 * 16 * 32];
  __shared__ float keybias[32];
  __shared__ int   kposA[32];

  const int tid  = threadIdx.x;
  const int lane = tid & 31;
  const int wave = tid >> 5;
  const int m = lane & 15, hi = lane >> 4;
  const int chunk = blockIdx.x >> 2;
  const int qsub  = blockIdx.x & 3;
  const int h     = blockIdx.y;
  const int qrow0 = chunk * 256 + qsub * 64 + wave * 16;
  const int hcol  = h * DH;

  FragB16 aq[2];
  for (int k0 = 0; k0 < 2; ++k0) {
    int qrow = qrow0 + m;
    for (int g = 0; g < 8; ++g) {
      int kk0 = ((g & 4) << 2) | ((g & 3) << 1) | (hi << 3);
      aq[k0].u[g] = *(const unsigned int*)&qkvb[(size_t)qrow * 2304 + hcol + k0 * 32 + kk0];
    }
  }

  v8f acc_o[4] = {};
  float mrow[8], lrow[8];
  for (int r = 0; r < 8; ++r) { mrow[r] = -3.0e38f; lrow[r] = 0.f; }

  const int kbase = chunk * 256 - 256;

  for (int t = 0; t < 25; ++t) {            // 24 window tiles + 1 global tile
    const bool isWin = (t < 24);
    if (tid < 32) {
      int kpos; float bias;
      if (isWin) {
        kpos = kbase + t * 32 + tid;
        bool ok = (kpos >= 0) && (kpos < SQ) && (amask[kpos] > 0) && (gmask[kpos] == 0);
        bias = ok ? 0.f : NEGF;
      } else {
        if (tid < NGT) { kpos = gidx[tid]; bias = (gvalid[tid] > 0.f) ? 0.f : NEGF; }
        else           { kpos = 0;         bias = NEGF; }
      }
      kposA[tid] = kpos; keybias[tid] = bias;
    }
    for (int i = tid; i < 32 * 32; i += 128) {   // K tile: dword copies
      int slot = i >> 5, dp = (i & 31) << 1;
      int kpos; bool inr;
      if (isWin) { kpos = kbase + t * 32 + slot; inr = (kpos >= 0) && (kpos < SQ); }
      else       { inr = slot < NGT; kpos = inr ? gidx[slot] : 0; }
      unsigned int kv = inr ? *(const unsigned int*)&qkvb[(size_t)kpos * 2304 + 768 + hcol + dp] : 0u;
      *(unsigned int*)&Kt[slot * 64 + dp] = kv;
    }
    for (int i = tid; i < 32 * 64; i += 128) {   // V tile: transposed staging
      int slot = i >> 6, d = i & 63;
      int kpos; bool inr;
      if (isWin) { kpos = kbase + t * 32 + slot; inr = (kpos >= 0) && (kpos < SQ); }
      else       { inr = slot < NGT; kpos = inr ? gidx[slot] : 0; }
      Vt[d * 32 + slot] = inr ? qkvb[(size_t)kpos * 2304 + 1536 + hcol + d] : (unsigned short)0;
    }
    __syncthreads();

    float sc[2][8];
    for (int nt = 0; nt < 2; ++nt) {
      v8f s = {};
      for (int k0 = 0; k0 < 2; ++k0) {
        FragB16 kb;
        int key = nt * 16 + m;
        for (int g = 0; g < 8; ++g)
          kb.u[g] = *(const unsigned int*)&Kt[key * 64 + k0 * 32 + (((g & 4) << 2) | ((g & 3) << 1) | (hi << 3))];
        s = __builtin_amdgcn_wmma_f32_16x16x32_bf16(
            false, aq[k0].v, false, kb.v, (short)0, s, false, false);
      }
      int slot = nt * 16 + m;
      float bias = keybias[slot];
      int kpos = kposA[slot];
      for (int r = 0; r < 8; ++r) {
        float val = s[r] * 0.125f + bias;
        if (isWin) {
          int qpos = qrow0 + r + 8 * hi;
          int dd = kpos - qpos;
          val = (dd < -WINW || dd > WINW) ? NEGF : val;
        }
        sc[nt][r] = val;
      }
    }

    for (int r = 0; r < 8; ++r) {
      float mloc = fmaxf(sc[0][r], sc[1][r]);
      for (int mm = 1; mm < 16; mm <<= 1) mloc = fmaxf(mloc, __shfl_xor(mloc, mm, 32));
      float mnew = fmaxf(mrow[r], mloc);
      float alpha = __expf(mrow[r] - mnew);
      mrow[r] = mnew;
      float p0 = __expf(sc[0][r] - mnew); if (sc[0][r] <= -1.0e8f) p0 = 0.f;
      float p1 = __expf(sc[1][r] - mnew); if (sc[1][r] <= -1.0e8f) p1 = 0.f;
      float rs = p0 + p1;
      for (int mm = 1; mm < 16; mm <<= 1) rs += __shfl_xor(rs, mm, 32);
      lrow[r] = lrow[r] * alpha + rs;
      for (int j = 0; j < 4; ++j) acc_o[j][r] = acc_o[j][r] * alpha;
      int rowm = r + 8 * hi;
      pl[wave * 512 + rowm * 32 + m]      = p0;
      pl[wave * 512 + rowm * 32 + 16 + m] = p1;
    }
    asm volatile("s_wait_dscnt 0x0" ::: "memory");   // cross-lane DS RAW

    FragB16 pa;
    for (int g = 0; g < 8; ++g) {
      int kk0 = ((g & 4) << 2) | ((g & 3) << 1) | (hi << 3);
      const float* pp = &pl[wave * 512 + m * 32 + kk0];
      pa.u[g] = packbf(pp[0], pp[1]);
    }
    for (int j = 0; j < 4; ++j) {
      FragB16 vb;
      for (int g = 0; g < 8; ++g)
        vb.u[g] = *(const unsigned int*)&Vt[(j * 16 + m) * 32 + (((g & 4) << 2) | ((g & 3) << 1) | (hi << 3))];
      acc_o[j] = __builtin_amdgcn_wmma_f32_16x16x32_bf16(
          false, pa.v, false, vb.v, (short)0, acc_o[j], false, false);
    }
    __syncthreads();
  }

  for (int j = 0; j < 4; ++j)
    for (int r = 0; r < 8; ++r) {
      int row = qrow0 + r + 8 * hi;
      float o = acc_o[j][r] / fmaxf(lrow[r], 1e-20f);
      atto_bf[(size_t)row * HID + hcol + j * 16 + m] = f2bf(o);
    }
}

// Global tokens attend over all S keys (192 rows/layer) — VALU path.
__global__ __launch_bounds__(256)
void attn_global_kernel(const unsigned short* __restrict__ qkvb, const int* __restrict__ amask,
                        const int* __restrict__ gidx, float* __restrict__ og) {
  __shared__ float qsh[64];
  __shared__ float sc[SQ];
  __shared__ float red[8];
  __shared__ float osh[256];
  const int tid = threadIdx.x;
  const int g = blockIdx.x, h = blockIdx.y;
  const int gpos = gidx[g];
  if (tid < 64) qsh[tid] = bf2f(qkvb[(size_t)gpos * 2304 + h * 64 + tid]) * 0.125f;
  __syncthreads();
  float lmax = -3.0e38f;
  for (int s = tid; s < SQ; s += 256) {
    float dot = 0.f;
    const unsigned short* kp = &qkvb[(size_t)s * 2304 + 768 + h * 64];
    for (int d = 0; d < 64; ++d) dot += qsh[d] * bf2f(kp[d]);
    float v = (amask[s] > 0) ? dot : NEGF;
    sc[s] = v;
    lmax = fmaxf(lmax, v);
  }
  for (int mm = 1; mm < 32; mm <<= 1) lmax = fmaxf(lmax, __shfl_xor(lmax, mm, 32));
  if ((tid & 31) == 0) red[tid >> 5] = lmax;
  __syncthreads();
  float M = -3.0e38f;
  for (int i = 0; i < 8; ++i) M = fmaxf(M, red[i]);
  float lsum = 0.f;
  for (int s = tid; s < SQ; s += 256) { float p = __expf(sc[s] - M); sc[s] = p; lsum += p; }
  for (int mm = 1; mm < 32; mm <<= 1) lsum += __shfl_xor(lsum, mm, 32);
  __syncthreads();
  if ((tid & 31) == 0) red[tid >> 5] = lsum;
  __syncthreads();
  float Ssum = 0.f;
  for (int i = 0; i < 8; ++i) Ssum += red[i];
  const int d = tid & 63, part = tid >> 6;
  float acc = 0.f;
  for (int s = part * 1024; s < (part + 1) * 1024; ++s)
    acc += sc[s] * bf2f(qkvb[(size_t)s * 2304 + 1536 + h * 64 + d]);
  osh[tid] = acc;
  __syncthreads();
  if (tid < 64)
    og[((size_t)h * NGT + g) * 64 + tid] =
        (osh[tid] + osh[64 + tid] + osh[128 + tid] + osh[192 + tid]) / fmaxf(Ssum, 1e-20f);
}

__global__ __launch_bounds__(256)
void topk_kernel(const int* __restrict__ gmask, int* __restrict__ gidx,
                 float* __restrict__ gvalid) {
  __shared__ float sc[SQ];
  __shared__ float redv[8];
  __shared__ int   redi[8];
  const int tid = threadIdx.x;
  for (int s = tid; s < SQ; s += 256) sc[s] = (float)(gmask[s] * 2 * SQ - s);
  __syncthreads();
  for (int it = 0; it < NGT; ++it) {
    float bv = -3.0e38f; int bi = 0;
    for (int s = tid; s < SQ; s += 256) { float v = sc[s]; if (v > bv) { bv = v; bi = s; } }
    for (int mm = 1; mm < 32; mm <<= 1) {
      float ov = __shfl_xor(bv, mm, 32);
      int   oi = __shfl_xor(bi, mm, 32);
      if (ov > bv || (ov == bv && oi < bi)) { bv = ov; bi = oi; }
    }
    if ((tid & 31) == 0) { redv[tid >> 5] = bv; redi[tid >> 5] = bi; }
    __syncthreads();
    if (tid == 0) {
      float bbv = redv[0]; int bbi = redi[0];
      for (int i = 1; i < 8; ++i)
        if (redv[i] > bbv || (redv[i] == bbv && redi[i] < bbi)) { bbv = redv[i]; bbi = redi[i]; }
      gidx[it] = bbi;
      gvalid[it] = (float)gmask[bbi];
      sc[bbi] = -3.0e38f;
    }
    __syncthreads();
  }
}

__global__ void scatter_global_kernel(const float* __restrict__ og, const int* __restrict__ gidx,
                                      const float* __restrict__ gvalid,
                                      unsigned short* __restrict__ atto_bf) {
  int i = blockIdx.x * 256 + threadIdx.x;
  if (i >= NHD * NGT * DH) return;
  int h = i / (NGT * DH);
  int rem = i % (NGT * DH);
  int g = rem / DH, d = rem % DH;
  if (gvalid[g] > 0.f)
    atto_bf[(size_t)gidx[g] * HID + h * DH + d] = f2bf(og[i]);
}

__device__ __forceinline__ float block_sum256(float v, float* red) {
  for (int mm = 1; mm < 32; mm <<= 1) v += __shfl_xor(v, mm, 32);
  if ((threadIdx.x & 31) == 0) red[threadIdx.x >> 5] = v;
  __syncthreads();
  float s = 0.f;
  for (int i = 0; i < 8; ++i) s += red[i];
  __syncthreads();
  return s;
}

__global__ __launch_bounds__(256)
void embed_ln_kernel(const int* __restrict__ ids, const float* __restrict__ tok_emb,
                     const float* __restrict__ pos_emb, const float* __restrict__ g,
                     const float* __restrict__ b, float* __restrict__ h,
                     unsigned short* __restrict__ h_bf) {
  __shared__ float red[8];
  const int row = blockIdx.x, tid = threadIdx.x;
  const int id = ids[row];
  float x[3];
  for (int j = 0; j < 3; ++j) {
    int i = tid + j * 256;
    x[j] = tok_emb[(size_t)id * HID + i] + pos_emb[(size_t)row * HID + i];
  }
  float mu = block_sum256(x[0] + x[1] + x[2], red) / HID;
  float vs = 0.f;
  for (int j = 0; j < 3; ++j) { float d = x[j] - mu; vs += d * d; }
  float var = block_sum256(vs, red) / HID;
  float rstd = rsqrtf(var + 1e-5f);
  for (int j = 0; j < 3; ++j) {
    int i = tid + j * 256;
    float y = (x[j] - mu) * rstd * g[i] + b[i];
    h[(size_t)row * HID + i] = y;
    h_bf[(size_t)row * HID + i] = f2bf(y);
  }
}

__global__ __launch_bounds__(256)
void residual_ln_kernel(const float* __restrict__ resid, const float* __restrict__ f,
                        const float* __restrict__ g, const float* __restrict__ b,
                        float* __restrict__ out, unsigned short* __restrict__ out_bf) {
  __shared__ float red[8];
  const int row = blockIdx.x, tid = threadIdx.x;
  float x[3];
  for (int j = 0; j < 3; ++j) {
    int i = tid + j * 256;
    x[j] = resid[(size_t)row * HID + i] + f[(size_t)row * HID + i];
  }
  float mu = block_sum256(x[0] + x[1] + x[2], red) / HID;
  float vs = 0.f;
  for (int j = 0; j < 3; ++j) { float d = x[j] - mu; vs += d * d; }
  float var = block_sum256(vs, red) / HID;
  float rstd = rsqrtf(var + 1e-5f);
  for (int j = 0; j < 3; ++j) {
    int i = tid + j * 256;
    float y = (x[j] - mu) * rstd * g[i] + b[i];
    out[(size_t)row * HID + i] = y;
    out_bf[(size_t)row * HID + i] = f2bf(y);
  }
}

__global__ void pool_kernel(const float* __restrict__ h, const int* __restrict__ amask,
                            float* __restrict__ pooled) {
  int i = blockIdx.x * 256 + threadIdx.x;
  if (i >= HID) return;
  float s = 0.f, ms = 0.f;
  for (int t = 0; t < SQ; ++t) {
    float am = (amask[t] > 0) ? 1.f : 0.f;
    s += h[(size_t)t * HID + i] * am;
    ms += am;
  }
  pooled[i] = s / fmaxf(ms, 1.f);
}

__global__ __launch_bounds__(256)
void cls_kernel(const float* __restrict__ pooled, const float* __restrict__ W,
                const float* __restrict__ b, float* __restrict__ out) {
  __shared__ float red[8];
  const int tid = threadIdx.x;
  for (int c = 0; c < 2; ++c) {
    float s = 0.f;
    for (int i = tid; i < HID; i += 256) s += pooled[i] * W[i * 2 + c];
    float tot = block_sum256(s, red);
    if (tid == 0) out[c] = tot + b[c];
    __syncthreads();
  }
}

// ---------------------------------------------------------------------------
extern "C" void kernel_launch(void* const* d_in, const int* in_sizes, int n_in,
                              void* d_out, int out_size, void* d_ws, size_t ws_size,
                              hipStream_t stream) {
  const int*   ids     = (const int*)d_in[0];
  const int*   amask   = (const int*)d_in[1];
  const int*   gmask   = (const int*)d_in[2];
  const float* tok_emb = (const float*)d_in[3];
  const float* pos_emb = (const float*)d_in[4];
  const float* eln_w   = (const float*)d_in[5];
  const float* eln_b   = (const float*)d_in[6];
  const float* Wqkv    = (const float*)d_in[7];
  const float* bqkv    = (const float*)d_in[8];
  const float* Wo      = (const float*)d_in[9];
  const float* bo      = (const float*)d_in[10];
  const float* ln1w    = (const float*)d_in[11];
  const float* ln1b    = (const float*)d_in[12];
  const float* W1      = (const float*)d_in[13];
  const float* b1      = (const float*)d_in[14];
  const float* W2      = (const float*)d_in[15];
  const float* b2      = (const float*)d_in[16];
  const float* ln2w    = (const float*)d_in[17];
  const float* ln2b    = (const float*)d_in[18];
  const float* clsW    = (const float*)d_in[19];
  const float* clsb    = (const float*)d_in[20];
  float* out = (float*)d_out;

  float* ws = (float*)d_ws;
  size_t off = 0;
  float* h      = ws + off; off += (size_t)SQ * HID;
  float* proj   = ws + off; off += (size_t)SQ * HID;
  float* ffn2   = ws + off; off += (size_t)SQ * HID;
  float* og     = ws + off; off += (size_t)NHD * NGT * DH;
  float* pooled = ws + off; off += HID;
  float* gval   = ws + off; off += NGT;
  int*   gidx   = (int*)(ws + off); off += NGT;

  unsigned short* us = (unsigned short*)(ws + off);
  size_t uo = 0;
  unsigned short* h_bf    = us + uo; uo += (size_t)SQ * HID;
  unsigned short* qkv_bf  = us + uo; uo += (size_t)SQ * 3 * HID;
  unsigned short* atto_bf = us + uo; uo += (size_t)SQ * HID;
  unsigned short* ffn1_bf = us + uo; uo += (size_t)SQ * FFD;
  unsigned short* wq_t    = us + uo; uo += (size_t)HID * 3 * HID;
  unsigned short* wo_t    = us + uo; uo += (size_t)HID * HID;
  unsigned short* w1_t    = us + uo; uo += (size_t)HID * FFD;
  unsigned short* w2_t    = us + uo; uo += (size_t)FFD * HID;

  const dim3 tb(32, 8);

  topk_kernel<<<1, 256, 0, stream>>>(gmask, gidx, gval);
  embed_ln_kernel<<<SQ, 256, 0, stream>>>(ids, tok_emb, pos_emb, eln_w, eln_b, h, h_bf);

  for (int l = 0; l < LAYERS; ++l) {
    // per-layer weight convert + transpose into bf16 [N][K] (L2-resident)
    transpose_f32_bf16_kernel<<<dim3(HID / 32, 3 * HID / 32), tb, 0, stream>>>(
        Wqkv + (size_t)l * HID * 3 * HID, wq_t, HID, 3 * HID);
    transpose_f32_bf16_kernel<<<dim3(HID / 32, HID / 32), tb, 0, stream>>>(
        Wo + (size_t)l * HID * HID, wo_t, HID, HID);
    transpose_f32_bf16_kernel<<<dim3(HID / 32, FFD / 32), tb, 0, stream>>>(
        W1 + (size_t)l * HID * FFD, w1_t, HID, FFD);
    transpose_f32_bf16_kernel<<<dim3(FFD / 32, HID / 32), tb, 0, stream>>>(
        W2 + (size_t)l * FFD * HID, w2_t, FFD, HID);

    gemm_bf16_kernel<<<dim3(3 * HID / 64, SQ / 128), 256, 0, stream>>>(
        h_bf, wq_t, bqkv + (size_t)l * 3 * HID, nullptr, qkv_bf, SQ, 3 * HID, HID, 0);
    attn_local_kernel<<<dim3((SQ / WINW) * 4, NHD), 128, 0, stream>>>(
        qkv_bf, amask, gmask, gidx, gval, atto_bf);
    attn_global_kernel<<<dim3(NGT, NHD), 256, 0, stream>>>(qkv_bf, amask, gidx, og);
    scatter_global_kernel<<<(NHD * NGT * DH + 255) / 256, 256, 0, stream>>>(
        og, gidx, gval, atto_bf);
    gemm_bf16_kernel<<<dim3(HID / 64, SQ / 128), 256, 0, stream>>>(
        atto_bf, wo_t, bo + (size_t)l * HID, proj, nullptr, SQ, HID, HID, 0);
    residual_ln_kernel<<<SQ, 256, 0, stream>>>(h, proj, ln1w + (size_t)l * HID,
                                               ln1b + (size_t)l * HID, h, h_bf);
    gemm_bf16_kernel<<<dim3(FFD / 64, SQ / 128), 256, 0, stream>>>(
        h_bf, w1_t, b1 + (size_t)l * FFD, nullptr, ffn1_bf, SQ, FFD, HID, 1);
    gemm_bf16_kernel<<<dim3(HID / 64, SQ / 128), 256, 0, stream>>>(
        ffn1_bf, w2_t, b2 + (size_t)l * HID, ffn2, nullptr, SQ, HID, FFD, 0);
    residual_ln_kernel<<<SQ, 256, 0, stream>>>(h, ffn2, ln2w + (size_t)l * HID,
                                               ln2b + (size_t)l * HID, h, h_bf);
  }

  pool_kernel<<<(HID + 255) / 256, 256, 0, stream>>>(h, amask, pooled);
  cls_kernel<<<1, 256, 0, stream>>>(pooled, clsW, clsb, out);
}
